// GraphAttention_36258113913294
// MI455X (gfx1250) — matrix operand order repaired and checked by reading, not compiled, not run on hardware
//
#include <hip/hip_runtime.h>
#include <hip/hip_bf16.h>
#include <math.h>
#include <stdint.h>

typedef __bf16 bf16;
typedef __attribute__((ext_vector_type(16))) __bf16 v16bf;
typedef __attribute__((ext_vector_type(8)))  __bf16 v8bf;
typedef __attribute__((ext_vector_type(8)))  float  v8f;

// Problem constants
static constexpr int Bn   = 256;
static constexpr int Nn   = 133;
static constexpr int Dn   = 1024;
static constexpr int Hn   = 8;
static constexpr int Cn   = 128;
static constexpr int HC   = Hn * Cn;        // 1024
static constexpr int Mrows = Bn * Nn;       // 34048 (divisible by 64)
static constexpr float SLOPE = 0.2f;

// ---------------------------------------------------------------------------
// CDNA5 async global->LDS copy (16B per lane) + async-counter wait.
// LDS byte address = low 32 bits of the flat shared pointer (ISA §10.2: the
// LDS aperture keeps the logical LDS offset in addr[31:0]).
// ---------------------------------------------------------------------------
__device__ __forceinline__ unsigned lds_off(const void* p) {
  return (unsigned)(uintptr_t)p;
}
__device__ __forceinline__ void async_copy_b128(unsigned lds_addr, const void* gptr) {
  asm volatile("global_load_async_to_lds_b128 %0, %1, off"
               :: "v"(lds_addr), "v"((unsigned long long)(uintptr_t)gptr)
               : "memory");
}
__device__ __forceinline__ void wait_async_le3() {
  asm volatile("s_wait_asynccnt 0x3" ::: "memory");
}
__device__ __forceinline__ void wait_async_0() {
  asm volatile("s_wait_asynccnt 0x0" ::: "memory");
}

// ---------------------------------------------------------------------------
// fp32 -> bf16 conversion (grid-stride)
// ---------------------------------------------------------------------------
__global__ __launch_bounds__(256)
void cvt_f32_bf16(const float* __restrict__ src, bf16* __restrict__ dst, long n) {
  long i = (long)blockIdx.x * blockDim.x + threadIdx.x;
  long stride = (long)gridDim.x * blockDim.x;
  for (; i < n; i += stride) dst[i] = (bf16)src[i];
}

// ---------------------------------------------------------------------------
// g = x @ W^T  via v_wmma_f32_16x16x32_bf16, double-buffered async LDS pipeline.
// Block tile: 64(M) x 128(N).  8 waves in 2(M) x 4(N); each wave: 2x2 WMMA tiles.
// ---------------------------------------------------------------------------
__global__ __launch_bounds__(256)
void gemm_xw_wmma(const bf16* __restrict__ xb, const bf16* __restrict__ wb,
                  bf16* __restrict__ gb) {
  __shared__ alignas(64) bf16 xs[2][64 * 32];    // 2 x 4 KB
  __shared__ alignas(64) bf16 wls[2][128 * 32];  // 2 x 8 KB
  const int t    = threadIdx.x;
  const int lane = t & 31;
  const int wave = t >> 5;
  const int wm   = wave & 1;     // M sub-block (32 rows)
  const int wn   = wave >> 1;    // N sub-block (32 cols)
  const int mBase = blockIdx.x * 64;
  const int nBase = blockIdx.y * 128;
  const int l16 = lane & 15;
  const int h16 = lane >> 4;

  // Per-thread staging coordinates (16B x-chunk, 2 x 16B W-chunks per panel)
  const int xrow = t >> 2, xkc = (t & 3) * 8;    // x tile 64x32
  const int wrow = t >> 1, wkc = (t & 1) * 16;   // W tile 128x32
  const bf16* xsrc = xb + (size_t)(mBase + xrow) * Dn + xkc;
  const bf16* wsrc = wb + (size_t)(nBase + wrow) * Dn + wkc;
  const unsigned xo[2] = { lds_off(&xs[0][xrow * 32 + xkc]),
                           lds_off(&xs[1][xrow * 32 + xkc]) };
  const unsigned wo[2] = { lds_off(&wls[0][wrow * 32 + wkc]),
                           lds_off(&wls[1][wrow * 32 + wkc]) };

  v8f acc[2][2] = {};

  // Prologue: panel 0 -> buffer 0
  async_copy_b128(xo[0], xsrc);
  async_copy_b128(wo[0], wsrc);
  async_copy_b128(wo[0] + 16, wsrc + 8);

  for (int k = 0; k < Dn / 32; ++k) {
    const int p = k & 1;
    if (k + 1 < Dn / 32) {
      const int kb = (k + 1) * 32;
      async_copy_b128(xo[1 - p], xsrc + kb);
      async_copy_b128(wo[1 - p], wsrc + kb);
      async_copy_b128(wo[1 - p] + 16, wsrc + kb + 8);
      wait_async_le3();                 // panel k landed; panel k+1 in flight
    } else {
      wait_async_0();
    }
    __syncthreads();                    // all waves' async data for buffer p visible

    // ---- build fragments per documented CDNA5 16-bit layouts ----
    v16bf afr[2], bfr[2];
#pragma unroll
    for (int mt = 0; mt < 2; ++mt) {
      const bf16* ptr = &xs[p][(wm * 32 + mt * 16 + l16) * 32 + h16 * 8];
      v8bf lo = *(const v8bf*)ptr;             // K = h16*8 .. +7
      v8bf hi = *(const v8bf*)(ptr + 16);      // K = 16 + h16*8 .. +7
      afr[mt] = __builtin_shufflevector(lo, hi, 0,1,2,3,4,5,6,7,8,9,10,11,12,13,14,15);
    }
#pragma unroll
    for (int nt = 0; nt < 2; ++nt) {
      const bf16* ptr = &wls[p][(wn * 32 + nt * 16 + l16) * 32 + h16 * 8];
      v8bf lo = *(const v8bf*)ptr;             // K = h16*8 .. +7
      v8bf hi = *(const v8bf*)(ptr + 16);      // K = 16 + h16*8 .. +7
      bfr[nt] = __builtin_shufflevector(lo, hi, 0,1,2,3,4,5,6,7,8,9,10,11,12,13,14,15);
    }
#pragma unroll
    for (int mt = 0; mt < 2; ++mt)
#pragma unroll
      for (int nt = 0; nt < 2; ++nt)
        acc[mt][nt] = __builtin_amdgcn_wmma_f32_16x16x32_bf16(
            false, afr[mt], false, bfr[nt], (short)0, acc[mt][nt], false, false);
    __syncthreads();                    // buffer p free for iteration k+2's async writes
  }

  // ---- store g as bf16 (C/D layout: lane<16 -> M=r, lane>=16 -> M=8+r) ----
#pragma unroll
  for (int mt = 0; mt < 2; ++mt)
#pragma unroll
    for (int nt = 0; nt < 2; ++nt) {
      const int col = nBase + wn * 32 + nt * 16 + l16;
#pragma unroll
      for (int r = 0; r < 8; ++r) {
        const int mrow = mBase + wm * 32 + mt * 16 + h16 * 8 + r;
        gb[(size_t)mrow * HC + col] = (bf16)acc[mt][nt][r];
      }
    }
}

// ---------------------------------------------------------------------------
// si = g . wr, sj = g . wl   (one wave per head, one block per (b,n) row)
// ---------------------------------------------------------------------------
__global__ __launch_bounds__(256)
void scores_kernel(const bf16* __restrict__ gb, const float* __restrict__ attn_w,
                   float* __restrict__ si, float* __restrict__ sj) {
  const int m    = blockIdx.x;        // 0..Mrows-1
  const int h    = threadIdx.x >> 5;  // head
  const int lane = threadIdx.x & 31;
  const bf16* grow = gb + (size_t)m * HC + h * Cn;
  float sl = 0.f, sr = 0.f;
#pragma unroll
  for (int q = 0; q < 4; ++q) {
    const int c = lane + 32 * q;
    const float gv = (float)grow[c];
    sl += gv * attn_w[c];        // wl
    sr += gv * attn_w[Cn + c];   // wr
  }
#pragma unroll
  for (int off = 16; off > 0; off >>= 1) {
    sl += __shfl_down(sl, off, 32);
    sr += __shfl_down(sr, off, 32);
  }
  if (lane == 0) {
    sj[(size_t)m * Hn + h] = sl;   // g . wl
    si[(size_t)m * Hn + h] = sr;   // g . wr
  }
}

// ---------------------------------------------------------------------------
// Per-column softmax stats: for each (b,j,h):  max_i e  and  1/sum_i exp(e-max)
// e[i] = adj[i,j] ? leaky_relu(si[b,i,h] + sj[b,j,h]) : -inf.  One wave / task.
// ---------------------------------------------------------------------------
__global__ __launch_bounds__(256)
void colstats_kernel(const float* __restrict__ si, const float* __restrict__ sj,
                     const float* __restrict__ adj,
                     float* __restrict__ cmax, float* __restrict__ cden) {
  const int task = blockIdx.x * 8 + (threadIdx.x >> 5);  // = (b*Nn + j)*Hn + h
  const int lane = threadIdx.x & 31;
  const int h  = task & 7;
  const int bj = task >> 3;        // b*Nn + j
  const int j  = bj % Nn;
  const int b  = bj / Nn;
  const float sjv = sj[(size_t)bj * Hn + h];

  float mx = -3.4e38f;
  for (int i = lane; i < Nn; i += 32) {
    if (adj[i * Nn + j] != 0.f) {
      float e = si[(size_t)(b * Nn + i) * Hn + h] + sjv;
      e = e > 0.f ? e : SLOPE * e;
      mx = fmaxf(mx, e);
    }
  }
#pragma unroll
  for (int off = 16; off > 0; off >>= 1) mx = fmaxf(mx, __shfl_xor(mx, off, 32));

  float s = 0.f;
  for (int i = lane; i < Nn; i += 32) {
    if (adj[i * Nn + j] != 0.f) {
      float e = si[(size_t)(b * Nn + i) * Hn + h] + sjv;
      e = e > 0.f ? e : SLOPE * e;
      s += __expf(e - mx);
    }
  }
#pragma unroll
  for (int off = 16; off > 0; off >>= 1) s += __shfl_xor(s, off, 32);

  if (lane == 0) {
    cmax[task] = mx;
    cden[task] = 1.f / s;
  }
}

// ---------------------------------------------------------------------------
// out[b,i,h,:] = sum_j a[b,i,j,h] * g[b,j,h,:]  via WMMA.
// Block = (b, h, i-tile of 16).  a-tile (16x32) computed on the fly into LDS;
// g panel (32 x 128) transpose-staged into LDS.  Wave w -> 16-column slice.
// ---------------------------------------------------------------------------
__global__ __launch_bounds__(256)
void aggregate_kernel(const bf16* __restrict__ gb, const float* __restrict__ si,
                      const float* __restrict__ sj, const float* __restrict__ adj,
                      const float* __restrict__ cmax, const float* __restrict__ cden,
                      float* __restrict__ out) {
  __shared__ alignas(64) bf16 As[16 * 32];    // 1 KB : a[i_local][jj]
  __shared__ alignas(64) bf16 Bs[128 * 32];   // 8 KB : Bs[c][jj] (transposed)
  const int t    = threadIdx.x;
  const int lane = t & 31;
  const int wave = t >> 5;                    // column group
  int blk = blockIdx.x;
  const int it = blk % 9;  blk /= 9;          // i tile (133 -> 9 tiles of 16)
  const int h  = blk % Hn;
  const int b  = blk / Hn;
  const int l16 = lane & 15;
  const int h16 = lane >> 4;

  v8f acc = {};

  for (int kb = 0; kb < 160; kb += 32) {      // j padded to 160
    // ---- compute a-tile into LDS ----
    {
      const int il  = t >> 4;                 // 0..15
      const int jj0 = (t & 15) * 2;
      const int i = it * 16 + il;
#pragma unroll
      for (int u = 0; u < 2; ++u) {
        const int jj = jj0 + u;
        const int j  = kb + jj;
        float av = 0.f;
        if (i < Nn && j < Nn && adj[i * Nn + j] != 0.f) {
          const size_t cidx = (size_t)(b * Nn + j) * Hn + h;
          float e = si[(size_t)(b * Nn + i) * Hn + h] + sj[cidx];
          e = e > 0.f ? e : SLOPE * e;
          av = __expf(e - cmax[cidx]) * cden[cidx];
        }
        As[il * 32 + jj] = (bf16)av;
      }
    }
    // ---- transpose-stage g panel ----
    {
      const int jj = t >> 3;                  // 0..31
      const int cc = (t & 7) * 16;            // 0..112
      const int j  = kb + jj;
      if (j < Nn) {
        const bf16* src = gb + (size_t)(b * Nn + j) * HC + h * Cn + cc;
        v8bf lo = *(const v8bf*)src;
        v8bf hi = *(const v8bf*)(src + 8);
#pragma unroll
        for (int q = 0; q < 8; ++q) Bs[(cc + q) * 32 + jj] = lo[q];
#pragma unroll
        for (int q = 0; q < 8; ++q) Bs[(cc + 8 + q) * 32 + jj] = hi[q];
      } else {
#pragma unroll
        for (int q = 0; q < 16; ++q) Bs[(cc + q) * 32 + jj] = (bf16)0.f;
      }
    }
    __syncthreads();

    v16bf afr, bfr;
    {
      const bf16* ptr = As + l16 * 32 + h16 * 8;
      v8bf lo = *(const v8bf*)ptr;
      v8bf hi = *(const v8bf*)(ptr + 16);
      afr = __builtin_shufflevector(lo, hi, 0,1,2,3,4,5,6,7,8,9,10,11,12,13,14,15);
    }
    {
      const bf16* ptr = Bs + (wave * 16 + l16) * 32 + h16 * 8;
      v8bf lo = *(const v8bf*)ptr;
      v8bf hi = *(const v8bf*)(ptr + 16);
      bfr = __builtin_shufflevector(lo, hi, 0,1,2,3,4,5,6,7,8,9,10,11,12,13,14,15);
    }
    acc = __builtin_amdgcn_wmma_f32_16x16x32_bf16(
        false, afr, false, bfr, (short)0, acc, false, false);
    __syncthreads();
  }

  const int c = h * Cn + wave * 16 + l16;
#pragma unroll
  for (int r = 0; r < 8; ++r) {
    const int i = it * 16 + h16 * 8 + r;
    if (i < Nn) out[(size_t)(b * Nn + i) * HC + c] = acc[r];
  }
}

// ---------------------------------------------------------------------------
extern "C" void kernel_launch(void* const* d_in, const int* in_sizes, int n_in,
                              void* d_out, int out_size, void* d_ws, size_t ws_size,
                              hipStream_t stream) {
  const float* x      = (const float*)d_in[0];
  const float* W      = (const float*)d_in[1];
  const float* attn_w = (const float*)d_in[2];
  const float* adj    = (const float*)d_in[3];
  float* out = (float*)d_out;

  char* ws = (char*)d_ws;
  constexpr size_t nX = (size_t)Mrows * Dn;   // 34,865,152 elements
  constexpr size_t nW = (size_t)HC * Dn;      //  1,048,576 elements
  bf16* xb = (bf16*)ws;                               // nX * 2 bytes
  bf16* wb = (bf16*)(ws + nX * 2);                    // nW * 2 bytes
  bf16* gb = (bf16*)(ws + nX * 2 + nW * 2);           // nX * 2 bytes
  float* si = (float*)(ws + nX * 4 + nW * 2);         // Mrows*Hn floats
  float* sj = si + (size_t)Mrows * Hn;
  float* cm = sj + (size_t)Mrows * Hn;
  float* cd = cm + (size_t)Mrows * Hn;

  cvt_f32_bf16<<<2048, 256, 0, stream>>>(x, xb, (long)nX);
  cvt_f32_bf16<<<512,  256, 0, stream>>>(W, wb, (long)nW);

  gemm_xw_wmma<<<dim3(Mrows / 64, HC / 128), 256, 0, stream>>>(xb, wb, gb);

  scores_kernel<<<Mrows, 256, 0, stream>>>(gb, attn_w, si, sj);

  colstats_kernel<<<(Bn * Nn * Hn) / 8, 256, 0, stream>>>(si, sj, adj, cm, cd);

  aggregate_kernel<<<Bn * Hn * 9, 256, 0, stream>>>(gb, si, sj, adj, cm, cd, out);
}
